// PReNet_67327907332540
// MI455X (gfx1250) — matrix operand order repaired and checked by reading, not compiled, not run on hardware
//
#include <hip/hip_runtime.h>

#define PI_F 3.14159f

typedef __attribute__((ext_vector_type(16))) __bf16 v16bf;
typedef __attribute__((ext_vector_type(8)))  float  v8f;

// ---------------- helpers ----------------

__device__ inline unsigned short f2bf(float f) {
    union { float f; unsigned u; } x; x.f = f;
    unsigned u = x.u;
    u += 0x7FFFu + ((u >> 16) & 1u);            // round-to-nearest-even
    return (unsigned short)(u >> 16);
}
__device__ inline unsigned pk2(float a, float b) {
    return (unsigned)f2bf(a) | ((unsigned)f2bf(b) << 16);
}

// storage permutation so an A fragment is LDS-contiguous per lane:
// lanes 0-15 read k {0..7,16..23} at positions 0..15; lanes 16-31 read {8..15,24..31} at 16..31
// Only swaps bits 3<->4: 4-aligned groups of 4 stay contiguous.
__device__ inline int apermute(int k) {
    return (k & 7) | ((k & 8) << 1) | ((k & 16) >> 1);
}

#define LDS_PITCH 48   // halfs; 96B row stride keeps 32B fragment alignment, spreads banks

__device__ __constant__ float c_avg[20] = {
    40.712f, 34.04f, 29.417f, 26.423f, 79.706f, 35.224f, 23.27f, 22.036f,
    75.669f, 41.975f, 26.329f, 24.504f, 75.358f, 39.978f, 23.513f, 22.647f,
    74.556f, 27.541f, 19.826f, 20.395f };

// ---------------- implicit-GEMM conv + BN (+res) (+ReLU), bf16 WMMA ----------------
// GEMM (swapped):  D[co][pix] = sum_k W[co][k] * Im2col[k][pix]
//   A = weights  (M-dim = Co; OIHW is exactly [co][k]; float4 loads + b64 LDS stores)
//   B = im2col   (N-dim = pixels; lanes hold consecutive pixels -> coalesced NCHW I/O;
//                 float4 pixel loads for the K=1/stride=1 bulk)
// 64(Co) x 64(pix) tile / 256-thread block; 8 waves x 2 WMMA per 32-wide K step.
// Guards removed where provable: Co%64==0, M%64==0 for every layer; Kdim%32==0 for K in {1,3};
// K==1 has pad==0 so im2col coords are always in-bounds.

struct ConvArgs {
    const float* in; const float* w; const float* bias;
    const float* g; const float* bb; const float* mn; const float* vr;
    const float* res;
    float* out;
    int N, Ci, Hi, Wi, Co, stride, pad, woSh, hoSh;
    int mode;   // 0: relu(bn)  1: bn only  2: relu(bn + res)
};

template<int KK>
__global__ __launch_bounds__(256)
void conv_bn_wmma(ConvArgs A)
{
    __shared__ __attribute__((aligned(32))) unsigned short Ws[64][LDS_PITCH]; // [co][k]
    __shared__ __attribute__((aligned(32))) unsigned short Xs[64][LDS_PITCH]; // [pix][k]

    const int tid  = threadIdx.x;
    const int wave = tid >> 5, lane = tid & 31;
    const int spSh = A.woSh + A.hoSh;
    const int Kdim = A.Ci * KK * KK;
    const int WoM  = (1 << A.woSh) - 1, HoM = (1 << A.hoSh) - 1;
    const int coBase = blockIdx.x * 64;
    const int pxBase = blockIdx.y * 64;
    const int mw = (wave & 3) * 16;    // Co sub-block
    const int nw = (wave >> 2) * 32;   // pixel sub-block

    v8f acc0 = {0.f,0.f,0.f,0.f,0.f,0.f,0.f,0.f};
    v8f acc1 = {0.f,0.f,0.f,0.f,0.f,0.f,0.f,0.f};

    for (int k0 = 0; k0 < Kdim; k0 += 32) {
        // weights -> Ws : float4 loads (k-contiguous), packed ds_store_b64
        #pragma unroll
        for (int g = tid; g < 64 * 8; g += 256) {
            int cl = g >> 3, kg = (g & 7) * 4;
            int k = k0 + kg;
            float4 v4;
            if (KK == 7 && k >= Kdim) v4 = make_float4(0.f, 0.f, 0.f, 0.f);   // Kdim%4==0
            else v4 = *(const float4*)&A.w[(long)(coBase + cl) * Kdim + k];
            *(uint2*)&Ws[cl][kg] = make_uint2(pk2(v4.x, v4.y), pk2(v4.z, v4.w));
        }
        // im2col -> Xs (identity k layout; B fragments read k = seg + e directly)
        if (KK == 1 && A.stride == 1) {
            // bulk path: 4 consecutive pixels per float4 (4 | Wo, groups never cross rows)
            #pragma unroll
            for (int g = tid; g < 16 * 32; g += 256) {
                int pg = (g & 15) * 4, kl = g >> 4;
                int p = pxBase + pg;
                int wo = p & WoM, ho = (p >> A.woSh) & HoM, nb = p >> spSh;
                const float4 v4 = *(const float4*)
                    &A.in[((long)(nb * A.Ci + k0 + kl) * A.Hi + ho) * A.Wi + wo];
                Xs[pg + 0][kl] = f2bf(v4.x);
                Xs[pg + 1][kl] = f2bf(v4.y);
                Xs[pg + 2][kl] = f2bf(v4.z);
                Xs[pg + 3][kl] = f2bf(v4.w);
            }
        } else {
            #pragma unroll
            for (int i = tid; i < 64 * 32; i += 256) {
                int pl = i & 63, kl = i >> 6;
                int p = pxBase + pl, k = k0 + kl;
                float val = 0.f;
                int wo = p & WoM, ho = (p >> A.woSh) & HoM, nb = p >> spSh;
                if (KK == 1) {
                    // pad==0, always in-bounds; k < Kdim guaranteed (Ci%32==0)
                    val = A.in[((long)(nb * A.Ci + k) * A.Hi + ho * A.stride) * A.Wi + wo * A.stride];
                } else {
                    int ci = k / (KK * KK);
                    int rem = k - ci * (KK * KK);
                    int kh = rem / KK, kw = rem - kh * KK;
                    int hi = ho * A.stride - A.pad + kh;
                    int wi = wo * A.stride - A.pad + kw;
                    bool ok = (hi >= 0 && hi < A.Hi && wi >= 0 && wi < A.Wi);
                    if (KK == 7) ok = ok && (k < Kdim);
                    if (ok) val = A.in[((long)(nb * A.Ci + ci) * A.Hi + hi) * A.Wi + wi];
                }
                Xs[pl][kl] = f2bf(val);
            }
        }
        __syncthreads();

        const int seg = (lane >> 4) * 16;
        v16bf a  = *(const v16bf*)&Ws[mw + (lane & 15)][seg];
        v16bf b0 = *(const v16bf*)&Xs[nw + (lane & 15)][seg];
        v16bf b1 = *(const v16bf*)&Xs[nw + 16 + (lane & 15)][seg];
        acc0 = __builtin_amdgcn_wmma_f32_16x16x32_bf16(false, a, false, b0, (short)0, acc0, false, false);
        acc1 = __builtin_amdgcn_wmma_f32_16x16x32_bf16(false, a, false, b1, (short)0, acc1, false, false);
        __syncthreads();
    }

    // epilogue: row = co (BN folded per-row), cols = consecutive pixels -> coalesced stores
    #pragma unroll
    for (int r = 0; r < 8; ++r) {
        int co = coBase + mw + r + (lane >> 4) * 8;
        float sc  = A.g[co] * rsqrtf(A.vr[co] + 1e-5f);
        float off = sc * (A.bias[co] - A.mn[co]) + A.bb[co];
        #pragma unroll
        for (int half = 0; half < 2; ++half) {
            int p = pxBase + nw + half * 16 + (lane & 15);
            float y = sc * (half ? acc1[r] : acc0[r]) + off;
            int wo = p & WoM, ho = (p >> A.woSh) & HoM, nb = p >> spSh;
            long o = (((long)nb * A.Co + co) << spSh) + (ho << A.woSh) + wo;
            if (A.mode == 2) y += A.res[o];
            if (A.mode != 1) y = fmaxf(y, 0.f);
            A.out[o] = y;
        }
    }
}

// ---------------- FC GEMM via WMMA: out[b][n] = bias[n] + sum_k in[b][k]*w[n][k] -----
// M = B = 64 exactly (one block in M); Kin % 32 == 0; only Nout may need a guard.

__global__ __launch_bounds__(256)
void fc_wmma(const float* __restrict__ in, const float* __restrict__ w,
             const float* __restrict__ bias, float* __restrict__ out,
             int Kin, int Nout)
{
    __shared__ __attribute__((aligned(32))) unsigned short As2[64][LDS_PITCH]; // [b][k]
    __shared__ __attribute__((aligned(32))) unsigned short Bs2[64][LDS_PITCH]; // [n][k] identity

    const int tid = threadIdx.x;
    const int wave = tid >> 5, lane = tid & 31;
    const int nBase = blockIdx.y * 64;
    const int mw = (wave & 3) * 16;
    const int nw = (wave >> 2) * 32;

    v8f acc0 = {0.f,0.f,0.f,0.f,0.f,0.f,0.f,0.f};
    v8f acc1 = {0.f,0.f,0.f,0.f,0.f,0.f,0.f,0.f};

    for (int k0 = 0; k0 < Kin; k0 += 32) {
        #pragma unroll
        for (int g = tid; g < 64 * 8; g += 256) {
            int ml = g >> 3, kg = (g & 7) * 4;
            const float4 v4 = *(const float4*)&in[(long)ml * Kin + k0 + kg];
            *(uint2*)&As2[ml][kg] = make_uint2(pk2(v4.x, v4.y), pk2(v4.z, v4.w));
        }
        #pragma unroll
        for (int g = tid; g < 64 * 8; g += 256) {
            int nl = g >> 3, kg = (g & 7) * 4;
            int n = nBase + nl;
            float4 v4 = make_float4(0.f, 0.f, 0.f, 0.f);
            if (n < Nout) v4 = *(const float4*)&w[(long)n * Kin + k0 + kg];
            *(uint2*)&Bs2[nl][kg] = make_uint2(pk2(v4.x, v4.y), pk2(v4.z, v4.w));
        }
        __syncthreads();

        const int seg = (lane >> 4) * 16;
        v16bf a  = *(const v16bf*)&As2[mw + (lane & 15)][seg];
        v16bf b0 = *(const v16bf*)&Bs2[nw + (lane & 15)][seg];
        v16bf b1 = *(const v16bf*)&Bs2[nw + 16 + (lane & 15)][seg];
        acc0 = __builtin_amdgcn_wmma_f32_16x16x32_bf16(false, a, false, b0, (short)0, acc0, false, false);
        acc1 = __builtin_amdgcn_wmma_f32_16x16x32_bf16(false, a, false, b1, (short)0, acc1, false, false);
        __syncthreads();
    }

    const int mrow = mw + (lane >> 4) * 8;
    #pragma unroll
    for (int half = 0; half < 2; ++half) {
        int n = nBase + nw + half * 16 + (lane & 15);
        v8f& acc = half ? acc1 : acc0;
        if (n < Nout) {
            float bs = bias[n];
            #pragma unroll
            for (int r = 0; r < 8; ++r)
                out[(long)(mrow + r) * Nout + n] = acc[r] + bs;
        }
    }
}

// ---------------- small kernels ----------------

// conv1 output [64,64,64,64] -> pad(0,1,0,1), maxpool k3 s2 -> [64,64,32,32]; dims hardcoded
__global__ void maxpool_pad(const float* __restrict__ in, float* __restrict__ out)
{
    int i = blockIdx.x * 256 + threadIdx.x;          // < 64*64*32*32
    int wo = i & 31, ho = (i >> 5) & 31, c = (i >> 10) & 63, n = i >> 16;
    int base = ((n << 6) + c) << 12;                  // (n*64+c)*64*64
    float m = -INFINITY;
    #pragma unroll
    for (int dy = 0; dy < 3; ++dy)
        #pragma unroll
        for (int dx = 0; dx < 3; ++dx) {
            int h = ho * 2 + dy, w = wo * 2 + dx;
            float v = (h < 64 && w < 64) ? in[base + (h << 6) + w] : 0.f;   // zero pad
            m = fmaxf(m, v);
        }
    out[i] = m;
}

__global__ void fc_small(const float* __restrict__ in, const float* __restrict__ w,
                         const float* __restrict__ bias, float* __restrict__ out,
                         int B, int Kin, int Nout, int relu_in, int act_out)
{
    int i = blockIdx.x * blockDim.x + threadIdx.x;
    if (i >= B * Nout) return;
    int b = i / Nout, o = i % Nout;
    float s = bias[o];
    for (int k = 0; k < Kin; ++k) {
        float x = in[b * Kin + k];
        if (relu_in) x = fmaxf(x, 0.f);
        s += x * w[o * Kin + k];
    }
    if (act_out == 1) s = 2.f / (1.f + expf(-s)) - 1.f;   // 2*sigmoid-1
    out[i] = s;
}

// ---------------- forward kinematics ----------------

__device__ inline void mm4(const float* A, const float* B, float* C) {
    #pragma unroll
    for (int i = 0; i < 4; ++i)
        #pragma unroll
        for (int j = 0; j < 4; ++j) {
            float s = 0.f;
            #pragma unroll
            for (int k = 0; k < 4; ++k) s += A[i*4+k] * B[k*4+j];
            C[i*4+j] = s;
        }
}
__device__ inline void zmat(float th, float* T) {
    float c = cosf(th), s = sinf(th);
    T[0]=c;  T[1]=-s; T[2]=0;  T[3]=0;
    T[4]=s;  T[5]=c;  T[6]=0;  T[7]=0;
    T[8]=0;  T[9]=0;  T[10]=1; T[11]=0;
    T[12]=0; T[13]=0; T[14]=0; T[15]=1;
}
__device__ inline void xmat(float al, float a, float* T) {
    float c = cosf(al), s = sinf(al);
    T[0]=1;  T[1]=0;  T[2]=0;   T[3]=a;
    T[4]=0;  T[5]=c;  T[6]=-s;  T[7]=0;
    T[8]=0;  T[9]=s;  T[10]=c;  T[11]=0;
    T[12]=0; T[13]=0; T[14]=0;  T[15]=1;
}
__device__ inline void stepT(float* T, float th, float al, float a) {
    float Z[16], X[16], t1[16];
    zmat(th, Z); xmat(al, a, X);
    mm4(T, Z, t1); mm4(t1, X, T);
}

__global__ void fk_kernel(const float* __restrict__ theta, const float* __restrict__ scale,
                          float* __restrict__ pos, int B)
{
    int b = blockIdx.x * blockDim.x + threadIdx.x;
    if (b >= B) return;
    const float* th = theta + b * 29;
    float a[20];
    #pragma unroll
    for (int i = 0; i < 20; ++i) a[i] = scale[b * 20 + i] * c_avg[i];

    const float NH = -PI_F / 2.f, PH = PI_F / 2.f;
    float T[16], Z[16], X[16];
    float* P = pos + b * 60;
    #define STPOS(idx) { P[(idx)*3+0]=T[3]; P[(idx)*3+1]=T[7]; P[(idx)*3+2]=T[11]; }

    zmat(th[0], Z); xmat(NH, 0.f, X); mm4(Z, X, T);
    stepT(T, th[1], PH, a[0]);  STPOS(0);
    stepT(T, th[2], NH, 0.f);
    stepT(T, th[3], PH, a[1]);  STPOS(1);
    stepT(T, th[4], NH, 0.f);
    stepT(T, th[5], PH, a[2]);  STPOS(2);
    stepT(T, th[6], NH, 0.f);
    stepT(T, th[7], 0.f, a[3]); STPOS(3);
    zmat(th[8], Z); xmat(0.f, a[4], X); mm4(Z, X, T); STPOS(4);
    stepT(T, th[9], NH, 0.f);
    for (int l = 0; l < 3; ++l) { stepT(T, th[10+l], 0.f, a[5+l]); STPOS(5+l); }
    zmat(PH, Z); xmat(0.f, a[8], X); mm4(Z, X, T); STPOS(8);
    stepT(T, th[13], NH, 0.f);
    for (int l = 0; l < 3; ++l) { stepT(T, th[14+l], 0.f, a[9+l]); STPOS(9+l); }
    zmat(th[17], Z); xmat(NH, 0.f, X); mm4(Z, X, T);
    stepT(T, th[18], PH, a[12]); STPOS(12);
    stepT(T, th[19], NH, 0.f);
    for (int l = 0; l < 3; ++l) { stepT(T, th[20+l], 0.f, a[13+l]); STPOS(13+l); }
    zmat(th[23], Z); xmat(NH, 0.f, X); mm4(Z, X, T);
    stepT(T, th[24], PH, a[16]); STPOS(16);
    stepT(T, th[25], NH, 0.f);
    for (int l = 0; l < 3; ++l) { stepT(T, th[26+l], 0.f, a[17+l]); STPOS(17+l); }
    #undef STPOS
}

__global__ void rot_kernel(const float* __restrict__ R, const float* __restrict__ pos,
                           float* __restrict__ y, int B)
{
    int i = blockIdx.x * blockDim.x + threadIdx.x;
    if (i >= B * 20 * 3) return;
    int ii = i % 3; int t = i / 3; int n = t % 20; int b = t / 20;
    const float* Rb = R + b * 9;
    const float* p  = pos + b * 60 + n * 3;
    y[i] = Rb[ii*3+0]*p[0] + Rb[ii*3+1]*p[1] + Rb[ii*3+2]*p[2];
}

// ---------------- host-side parameter mapping ----------------

struct BNp { const float *g, *b, *m, *v; };
struct CW  { const float *w, *b; BNp bn; };
struct Res { CW c1, c2, c3, cs; bool has_s; };
struct Params {
    const float* x;
    CW conv1, conv4e, conv4f;
    Res r2a, r2b, r2c, r3a, r3b, r3c, r4a, r4b, r4c, r4d;
    const float *t1w,*t1b,*t2w,*t2b,*s1w,*s1b,*s2w,*s2b,*v1w,*v1b,*v2w,*v2b,*dw,*db;
};

static void fill_insertion(void* const* din, Params& P) {
    int c = 0;
    auto nx = [&]() { return (const float*)din[c++]; };
    auto bn = [&](BNp& b) { b.g = nx(); b.b = nx(); b.m = nx(); b.v = nx(); };
    auto cw = [&](CW& w)  { w.w = nx(); w.b = nx(); bn(w.bn); };
    auto rs = [&](Res& r, bool sc) { cw(r.c1); cw(r.c2); cw(r.c3); r.has_s = sc; if (sc) cw(r.cs); };
    P.x = nx();
    cw(P.conv1);
    rs(P.r2a, true); rs(P.r2b, false); rs(P.r2c, false);
    rs(P.r3a, true); rs(P.r3b, false); rs(P.r3c, false);
    rs(P.r4a, true); rs(P.r4b, false); rs(P.r4c, false); rs(P.r4d, false);
    cw(P.conv4e); cw(P.conv4f);
    P.t1w = nx(); P.t1b = nx(); P.t2w = nx(); P.t2b = nx();
    P.s1w = nx(); P.s1b = nx(); P.s2w = nx(); P.s2b = nx();
    P.v1w = nx(); P.v1b = nx(); P.v2w = nx(); P.v2b = nx();
    P.dw  = nx(); P.db  = nx();
}

// jax tree_flatten sorts dict keys: 'params' < 'x'; inside dicts alphabetical.
static void fill_sorted(void* const* din, Params& P) {
    int c = 0;
    auto nx = [&]() { return (const float*)din[c++]; };
    auto bn = [&](BNp& b) { b.g = nx(); b.b = nx(); b.m = nx(); b.v = nx(); };
    auto cw = [&](CW& w)  { w.b = nx(); bn(w.bn); w.w = nx(); };           // b < bn < w
    auto rs = [&](Res& r, bool sc) {                                       // b1<b2<b3<bn1<bn2<bn3<bns<bs<w1<w2<w3<ws
        r.c1.b = nx(); r.c2.b = nx(); r.c3.b = nx();
        bn(r.c1.bn); bn(r.c2.bn); bn(r.c3.bn);
        if (sc) { bn(r.cs.bn); r.cs.b = nx(); }
        r.c1.w = nx(); r.c2.w = nx(); r.c3.w = nx();
        if (sc) r.cs.w = nx();
        r.has_s = sc;
    };
    cw(P.conv1); cw(P.conv4e); cw(P.conv4f);
    P.db = nx(); P.dw = nx();
    P.v1b = nx(); P.v1w = nx(); P.v2b = nx(); P.v2w = nx();
    P.s1b = nx(); P.s1w = nx(); P.s2b = nx(); P.s2w = nx();
    P.t1b = nx(); P.t1w = nx(); P.t2b = nx(); P.t2w = nx();
    rs(P.r2a, true); rs(P.r2b, false); rs(P.r2c, false);
    rs(P.r3a, true); rs(P.r3b, false); rs(P.r3c, false);
    rs(P.r4a, true); rs(P.r4b, false); rs(P.r4c, false); rs(P.r4d, false);
    P.x = nx();
}

static int ilog2(int v) { int s = 0; while ((1 << s) < v) ++s; return s; }

static void convL(hipStream_t s, const float* in, const CW& cw, const float* res,
                  float* out, int N, int Ci, int Hi, int Wi, int Co,
                  int K, int stride, int pad, int mode)
{
    int Ho = (Hi + 2 * pad - K) / stride + 1;
    int Wo = (Wi + 2 * pad - K) / stride + 1;
    ConvArgs A;
    A.in = in; A.w = cw.w; A.bias = cw.b;
    A.g = cw.bn.g; A.bb = cw.bn.b; A.mn = cw.bn.m; A.vr = cw.bn.v;
    A.res = res; A.out = out;
    A.N = N; A.Ci = Ci; A.Hi = Hi; A.Wi = Wi; A.Co = Co;
    A.stride = stride; A.pad = pad;
    A.woSh = ilog2(Wo); A.hoSh = ilog2(Ho);
    A.mode = mode;
    int M = N * Ho * Wo;
    dim3 grid(Co / 64, M / 64);          // both exact multiples for every layer
    if (K == 1)      conv_bn_wmma<1><<<grid, 256, 0, s>>>(A);
    else if (K == 3) conv_bn_wmma<3><<<grid, 256, 0, s>>>(A);
    else             conv_bn_wmma<7><<<grid, 256, 0, s>>>(A);
}

static void resblock(hipStream_t s, const Res& R, const float* x, float* out,
                     float* m1, float* m2, float* sbuf,
                     int N, int Ci, int Hi, int Wi, int Cm, int Co, int stride)
{
    int Ho = (Hi - 1) / stride + 1, Wo = (Wi - 1) / stride + 1;
    const float* resid = x;
    if (R.has_s) { convL(s, x, R.cs, nullptr, sbuf, N, Ci, Hi, Wi, Co, 1, stride, 0, 1); resid = sbuf; }
    convL(s, x,  R.c1, nullptr, m1, N, Ci, Hi, Wi, Cm, 1, stride, 0, 0);
    convL(s, m1, R.c2, nullptr, m2, N, Cm, Ho, Wo, Cm, 3, 1, 1, 0);
    convL(s, m2, R.c3, resid, out,  N, Cm, Ho, Wo, Co, 1, 1, 0, 2);
}

// ---------------- entry ----------------

extern "C" void kernel_launch(void* const* d_in, const int* in_sizes, int n_in,
                              void* d_out, int out_size, void* d_ws, size_t ws_size,
                              hipStream_t stream)
{
    (void)n_in; (void)out_size; (void)ws_size;
    Params P;
    if (in_sizes[0] == 64 * 4 * 128 * 128) fill_insertion(d_in, P);
    else                                   fill_sorted(d_in, P);

    float* ws = (float*)d_ws;
    float* A   = ws;
    float* Bb  = ws + 17000000L;
    float* S   = ws + 34000000L;
    float* M1  = ws + 51000000L;
    float* M2  = ws + 56000000L;
    float* T1  = ws + 61000000L;   // [64,256]
    float* V1  = T1 + 16384;       // [64,256]
    float* SC1 = V1 + 16384;       // [64,3]
    float* T2  = SC1 + 256;        // [64,7]

    float* out     = (float*)d_out;
    float* o_y     = out;          // [64,20,3]
    float* o_scale = out + 3840;   // [64,20]
    float* o_theta = out + 5120;   // [64,29]
    float* o_R     = out + 6976;   // [64,3,3]
    float* o_pos   = out + 7552;   // [64,20,3]

    // backbone
    convL(stream, P.x, P.conv1, nullptr, Bb, 64, 4, 128, 128, 64, 7, 2, 3, 0);   // -> [64,64,64,64]
    maxpool_pad<<<16384, 256, 0, stream>>>(Bb, A);                               // -> [64,64,32,32]
    resblock(stream, P.r2a, A,  Bb, M1, M2, S, 64,  64, 32, 32,  64,  256, 1);
    resblock(stream, P.r2b, Bb, A,  M1, M2, S, 64, 256, 32, 32,  64,  256, 1);
    resblock(stream, P.r2c, A,  Bb, M1, M2, S, 64, 256, 32, 32,  64,  256, 1);
    resblock(stream, P.r3a, Bb, A,  M1, M2, S, 64, 256, 32, 32, 128,  512, 2);
    resblock(stream, P.r3b, A,  Bb, M1, M2, S, 64, 512, 16, 16, 128,  512, 1);
    resblock(stream, P.r3c, Bb, A,  M1, M2, S, 64, 512, 16, 16, 128,  512, 1);
    resblock(stream, P.r4a, A,  Bb, M1, M2, S, 64, 512, 16, 16, 256, 1024, 2);
    resblock(stream, P.r4b, Bb, A,  M1, M2, S, 64, 1024, 8, 8, 256, 1024, 1);
    resblock(stream, P.r4c, A,  Bb, M1, M2, S, 64, 1024, 8, 8, 256, 1024, 1);
    resblock(stream, P.r4d, Bb, A,  M1, M2, S, 64, 1024, 8, 8, 256, 1024, 1);
    convL(stream, A,  P.conv4e, nullptr, Bb, 64, 1024, 8, 8, 512, 3, 1, 1, 0);
    convL(stream, Bb, P.conv4f, nullptr, A,  64,  512, 8, 8, 256, 3, 1, 1, 0);
    // A == flat features [64, 16384]

    // heads (16384-K GEMMs on WMMA; tiny ones on VALU)
    fc_wmma<<<dim3(1, 4), 256, 0, stream>>>(A, P.t1w, P.t1b, T1, 16384, 256);
    fc_small<<<(64 * 7 + 255) / 256, 256, 0, stream>>>(T1, P.t2w, P.t2b, T2, 64, 256, 7, 1, 0);
    fc_small<<<(64 * 29 + 255) / 256, 256, 0, stream>>>(T2, P.dw, P.db, o_theta, 64, 7, 29, 0, 0);

    fc_wmma<<<dim3(1, 1), 256, 0, stream>>>(A, P.s1w, P.s1b, SC1, 16384, 3);
    fc_small<<<(64 * 20 + 255) / 256, 256, 0, stream>>>(SC1, P.s2w, P.s2b, o_scale, 64, 3, 20, 0, 0);

    fc_wmma<<<dim3(1, 4), 256, 0, stream>>>(A, P.v1w, P.v1b, V1, 16384, 256);
    fc_small<<<(64 * 9 + 255) / 256, 256, 0, stream>>>(V1, P.v2w, P.v2b, o_R, 64, 256, 9, 0, 1);

    // kinematics + rotation
    fk_kernel<<<1, 64, 0, stream>>>(o_theta, o_scale, o_pos, 64);
    rot_kernel<<<(64 * 60 + 255) / 256, 256, 0, stream>>>(o_R, o_pos, o_y, 64);
}
